// CCL_80161269613141
// MI455X (gfx1250) — compile-verified, matrix operand order checked by reading, not run on hardware
//
#include <hip/hip_runtime.h>
#include <math.h>

#define N 8192
#define KSEL 4096
#define EPSF 1e-10f
#define MASK_WORDS (N / 32)          // 256 uint32 words per row
#define NCHUNK 8
#define ROWS_PER_CHUNK (N / NCHUNK)  // 1024
#define TILE 64

typedef __attribute__((ext_vector_type(2)))  float    v2f;
typedef __attribute__((ext_vector_type(8)))  float    v8f;
typedef __attribute__((ext_vector_type(16))) _Float16 v16h;

// ---------------------------------------------------------------------------
// Kernel 1: row sums of s = exp(2*scores). One block per row, coalesced.
// ---------------------------------------------------------------------------
__global__ void k_rowsum(const float* __restrict__ sc, float* __restrict__ rowsum) {
    int row = blockIdx.x;
    const float* p = sc + (size_t)row * N;
    float acc = 0.0f;
    for (int j = threadIdx.x; j < N; j += 256)
        acc += __expf(2.0f * p[j]);
    __shared__ float red[256];
    red[threadIdx.x] = acc;
    __syncthreads();
    for (int o = 128; o > 0; o >>= 1) {
        if (threadIdx.x < o) red[threadIdx.x] += red[threadIdx.x + o];
        __syncthreads();
    }
    if (threadIdx.x == 0) rowsum[row] = red[0];
}

// ---------------------------------------------------------------------------
// Kernel 2: column sums via WMMA. A = ones => D[m][n] = sum_k B[k][n] + C.
// The ones-operand makes the result invariant to the K-permutation of the
// B VGPR layout; the only assumption used is n == lane%16.
// One wave per block; strip = 16 columns, chunk = 1024 rows.
// ---------------------------------------------------------------------------
__global__ void k_colsum_wmma(const float* __restrict__ sc, float* __restrict__ part) {
    int strip = blockIdx.x & 511;   // 512 strips of 16 columns
    int chunk = blockIdx.x >> 9;    // 8 row chunks
    int lane  = threadIdx.x;        // wave32
    int col   = strip * 16 + (lane & 15);
    int r0    = chunk * ROWS_PER_CHUNK;

#if __has_builtin(__builtin_amdgcn_wmma_f32_16x16x4_f32)
    v8f acc = {};
    v2f ones; ones.x = 1.0f; ones.y = 1.0f;
    int rsel = lane >> 4;           // which of the two rows in each B VGPR
    for (int r = r0; r < r0 + ROWS_PER_CHUNK; r += 4) {
        v2f b;
        b.x = __expf(2.0f * sc[(size_t)(r + rsel)     * N + col]);
        b.y = __expf(2.0f * sc[(size_t)(r + 2 + rsel) * N + col]);
        acc = __builtin_amdgcn_wmma_f32_16x16x4_f32(
            false, ones, false, b, (short)0, acc, false, false);
    }
    float csum = acc[0];
#else
    // Fallback: codegen-confirmed f16 WMMA, same ones-trick (K=32 per step).
    v8f acc = {};
    v16h onesh;
#pragma unroll
    for (int t = 0; t < 16; ++t) onesh[t] = (_Float16)1.0f;
    int par = lane >> 4;            // lanes 0-15: even rows, 16-31: odd rows
    for (int r = r0; r < r0 + ROWS_PER_CHUNK; r += 32) {
        v16h b;
#pragma unroll
        for (int t = 0; t < 16; ++t)
            b[t] = (_Float16)__expf(2.0f * sc[(size_t)(r + 2 * t + par) * N + col]);
        acc = __builtin_amdgcn_wmma_f32_16x16x32_f16(
            false, onesh, false, b, (short)0, acc, false, false);
    }
    float csum = acc[0];
#endif
    if (lane < 16)
        part[(size_t)chunk * N + strip * 16 + lane] = csum;
}

__global__ void k_colsum_reduce(const float* __restrict__ part, float* __restrict__ colsum) {
    int j = blockIdx.x * 256 + threadIdx.x;
    float s = 0.0f;
    for (int c = 0; c < NCHUNK; ++c) s += part[(size_t)c * N + j];
    colsum[j] = s;
}

// ---------------------------------------------------------------------------
// Kernel 3: exact per-row top-4096 mask via LDS radix select.
// Diagonal keyed to 0 (below every real value) == "rowmin - 1" semantics.
// Ties at the threshold resolved in index order (matches top_k tie-break).
// ---------------------------------------------------------------------------
__global__ void k_mask(const float* __restrict__ randn, unsigned* __restrict__ maskbits) {
    int row = blockIdx.x;
    int tid = threadIdx.x;          // 256 threads
    __shared__ unsigned keys[N];    // 32 KB
    __shared__ unsigned hist[256];
    __shared__ unsigned scan[256];
    __shared__ unsigned bcast[2];

    const float* p = randn + (size_t)row * N;
    for (int j = tid; j < N; j += 256) {
        unsigned bits = __float_as_uint(p[j]);
        unsigned k = (bits & 0x80000000u) ? ~bits : (bits | 0x80000000u); // order-preserving
        if (j == row) k = 0u;       // force diagonal below everything
        keys[j] = k;
    }
    __syncthreads();

    unsigned pref = 0u, prefmask = 0u, krem = KSEL;
    for (int pass = 3; pass >= 0; --pass) {
        int sh = pass * 8;
        hist[tid] = 0u;
        __syncthreads();
        for (int j = tid; j < N; j += 256) {
            unsigned k = keys[j];
            if ((k & prefmask) == pref)
                atomicAdd(&hist[(k >> sh) & 255u], 1u);
        }
        __syncthreads();
        if (tid == 0) {
            unsigned cum = 0u; int d = 0;
            for (int b = 255; b >= 0; --b) {
                unsigned c = hist[b];
                if (cum + c >= krem) { d = b; break; }
                cum += c;
            }
            bcast[0] = (unsigned)d;
            bcast[1] = krem - cum;  // quota remaining inside chosen bin
        }
        __syncthreads();
        unsigned d = bcast[0];
        krem = bcast[1];
        pref |= d << sh;
        prefmask |= 255u << sh;
        __syncthreads();
    }
    unsigned T = pref;              // threshold key (the 4096-th largest)
    unsigned quota = krem;          // how many ==T to take, earliest index first

    int base = tid * 32;            // contiguous 32-element chunk per thread
    unsigned ceq = 0u;
    for (int b = 0; b < 32; ++b) ceq += (keys[base + b] == T) ? 1u : 0u;
    scan[tid] = ceq;
    __syncthreads();
    for (int off = 1; off < 256; off <<= 1) {
        unsigned v = (tid >= off) ? scan[tid - off] : 0u;
        __syncthreads();
        scan[tid] += v;
        __syncthreads();
    }
    unsigned run = scan[tid] - ceq; // exclusive prefix count of equals
    unsigned bits = 0u;
    for (int b = 0; b < 32; ++b) {
        unsigned k = keys[base + b];
        bool take = (k > T) || (k == T && run < quota);
        if (k == T) run++;
        bits |= (take ? 1u : 0u) << b;
    }
    maskbits[(size_t)row * MASK_WORDS + tid] = bits;
}

// ---------------------------------------------------------------------------
// Kernel 4: masked log terms over 64x64 tiles (direct + transposed tile in LDS).
// term(i,j) = log(1 - s[i][j]/(rowsum[i]+eps) + eps)
//           + log(1 - s[j][i]/(colsum[i]+eps) + eps)
// ---------------------------------------------------------------------------
__global__ void k_terms(const float* __restrict__ sc,
                        const float* __restrict__ rowsum,
                        const float* __restrict__ colsum,
                        const unsigned* __restrict__ maskbits,
                        float* __restrict__ blocksums) {
    __shared__ float tA[TILE][TILE + 1];
    __shared__ float tB[TILE][TILE + 1];
    __shared__ float rs[TILE];
    __shared__ float cs[TILE];
    __shared__ unsigned mw[TILE * 2];
    __shared__ float red[256];

    int i0 = blockIdx.y * TILE, j0 = blockIdx.x * TILE;
    int tid = threadIdx.x;          // 256 threads

    for (int idx = tid; idx < TILE * TILE; idx += 256) {
        int r = idx >> 6, c = idx & 63;
        tA[r][c] = sc[(size_t)(i0 + r) * N + (j0 + c)];  // direct tile
        tB[r][c] = sc[(size_t)(j0 + r) * N + (i0 + c)];  // transposed tile
    }
    if (tid < TILE) { rs[tid] = rowsum[i0 + tid]; cs[tid] = colsum[i0 + tid]; }
    if (tid < TILE * 2) {
        int r = tid >> 1, w = tid & 1;
        mw[tid] = maskbits[(size_t)(i0 + r) * MASK_WORDS + (j0 >> 5) + w];
    }
    __syncthreads();

    float acc = 0.0f;
    for (int kk = 0; kk < 16; ++kk) {
        int idx = tid + kk * 256;
        int ii = idx >> 6, jj = idx & 63;
        unsigned bit = (mw[ii * 2 + (jj >> 5)] >> (jj & 31)) & 1u;
        if (bit) {
            float s1 = __expf(2.0f * tA[ii][jj]);
            float x1 = 1.0f - s1 / (rs[ii] + EPSF) + EPSF;
            float s2 = __expf(2.0f * tB[jj][ii]);
            float x2 = 1.0f - s2 / (cs[ii] + EPSF) + EPSF;
            acc += __logf(x1) + __logf(x2);
        }
    }
    red[tid] = acc;
    __syncthreads();
    for (int o = 128; o > 0; o >>= 1) {
        if (tid < o) red[tid] += red[tid + o];
        __syncthreads();
    }
    if (tid == 0) blocksums[blockIdx.y * gridDim.x + blockIdx.x] = red[0];
}

__global__ void k_finalize(const float* __restrict__ blocksums, int nblocks,
                           float* __restrict__ out) {
    __shared__ float red[256];
    float acc = 0.0f;
    for (int i = threadIdx.x; i < nblocks; i += 256) acc += blocksums[i];
    red[threadIdx.x] = acc;
    __syncthreads();
    for (int o = 128; o > 0; o >>= 1) {
        if (threadIdx.x < o) red[threadIdx.x] += red[threadIdx.x + o];
        __syncthreads();
    }
    if (threadIdx.x == 0) out[0] = -red[0] / (float)N;
}

// ---------------------------------------------------------------------------
extern "C" void kernel_launch(void* const* d_in, const int* in_sizes, int n_in,
                              void* d_out, int out_size, void* d_ws, size_t ws_size,
                              hipStream_t stream) {
    const float* scores = (const float*)d_in[0];
    const float* randn  = (const float*)d_in[1];
    float* out = (float*)d_out;

    // Workspace layout (all regions fully overwritten each call; ~8.8 MB):
    float*    rowsum    = (float*)d_ws;                                  // N
    float*    colsum    = rowsum + N;                                    // N
    float*    colpart   = colsum + N;                                    // NCHUNK*N
    unsigned* maskbits  = (unsigned*)(colpart + (size_t)NCHUNK * N);     // N*MASK_WORDS
    float*    blocksums = (float*)(maskbits + (size_t)N * MASK_WORDS);   // (N/TILE)^2

    k_rowsum<<<N, 256, 0, stream>>>(scores, rowsum);
    k_colsum_wmma<<<(N / 16) * NCHUNK, 32, 0, stream>>>(scores, colpart);
    k_colsum_reduce<<<N / 256, 256, 0, stream>>>(colpart, colsum);
    k_mask<<<N, 256, 0, stream>>>(randn, maskbits);
    dim3 grid(N / TILE, N / TILE);
    k_terms<<<grid, 256, 0, stream>>>(scores, rowsum, colsum, maskbits, blocksums);
    k_finalize<<<1, 256, 0, stream>>>(blocksums, (N / TILE) * (N / TILE), out);
}
